// PerspectiveNetwork_5411658793655
// MI455X (gfx1250) — compile-verified, matrix operand order checked by reading, not run on hardware
//
#include <hip/hip_runtime.h>

typedef __attribute__((ext_vector_type(16))) _Float16 v16h;
typedef __attribute__((ext_vector_type(8)))  _Float16 h8;
typedef __attribute__((ext_vector_type(4)))  _Float16 h4;
typedef __attribute__((ext_vector_type(8)))  float    v8f;

#define BATCH 16384
#define NF    768
#define FT    1024
#define BM    128
#define BN    128
#define BK    32
#define LDB   40   // padded row stride in halves (32 + 8)

// ---------------------------------------------------------------------------
// Main fused kernel: feature-transformer GEMM (WMMA f16 -> f32), SCReLU,
// output-layer partial dot product, atomic accumulation into out_acc[b].
// grid = (FT/BN, BATCH/BM, 2 perspectives), block = 256 threads (8 wave32).
// ---------------------------------------------------------------------------
__global__ __launch_bounds__(256) void ft_gemm_fused(
    const float* __restrict__ board_stm,
    const float* __restrict__ board_nstm,
    const float* __restrict__ W_ft,
    const float* __restrict__ b_ft,
    const float* __restrict__ W_out,
    float* __restrict__ out_acc)
{
    __shared__ _Float16 ldsA[BM * LDB];
    __shared__ _Float16 ldsB[BN * LDB];

    const int tid   = threadIdx.x;
    const int lane  = tid & 31;
    const int wid   = tid >> 5;      // 0..7
    const int wm    = wid & 3;       // wave row  -> rows  wm*32 .. +32
    const int wn    = wid >> 2;      // wave col  -> cols  wn*64 .. +64
    const int lr    = lane & 15;     // row/col within 16x16 tile
    const int lhalf = lane >> 4;     // 0 or 1
    const int klo   = lhalf * 8;     // K sub-offset per ISA f16 A/B layout

    const int n0 = blockIdx.x * BN;
    const int m0 = blockIdx.y * BM;
    const int p  = blockIdx.z;       // 0 = stm, 1 = nstm

    const float* __restrict__ boardP = p ? board_nstm : board_stm;

    v8f c[2][4];
    #pragma unroll
    for (int mi = 0; mi < 2; ++mi)
        #pragma unroll
        for (int t = 0; t < 4; ++t)
            #pragma unroll
            for (int j = 0; j < 8; ++j) c[mi][t][j] = 0.0f;

    for (int k0 = 0; k0 < NF; k0 += BK) {
        // ---- stage A tile: 128 rows x 32 feats, f32 -> f16 ----
        #pragma unroll
        for (int it = 0; it < 4; ++it) {
            int idx = tid + it * 256;          // 0..1023 float4 slots
            int row = idx >> 3;
            int c4  = idx & 7;
            const float4 v = *(const float4*)(boardP + (size_t)(m0 + row) * NF + k0 + c4 * 4);
            h4 hv = { (_Float16)v.x, (_Float16)v.y, (_Float16)v.z, (_Float16)v.w };
            *(h4*)&ldsA[row * LDB + c4 * 4] = hv;
        }
        // ---- stage B tile: W_ft rows n0..n0+127, same 32 feats ----
        #pragma unroll
        for (int it = 0; it < 4; ++it) {
            int idx = tid + it * 256;
            int row = idx >> 3;
            int c4  = idx & 7;
            const float4 v = *(const float4*)(W_ft + (size_t)(n0 + row) * NF + k0 + c4 * 4);
            h4 hv = { (_Float16)v.x, (_Float16)v.y, (_Float16)v.z, (_Float16)v.w };
            *(h4*)&ldsB[row * LDB + c4 * 4] = hv;
        }
        __syncthreads();

        // ---- load WMMA fragments from LDS (two b128 chunks per lane) ----
        v16h a[2], b[4];
        #pragma unroll
        for (int mi = 0; mi < 2; ++mi) {
            int off = (wm * 32 + mi * 16 + lr) * LDB + klo;
            h8 lo = *(const h8*)&ldsA[off];
            h8 hi = *(const h8*)&ldsA[off + 16];
            #pragma unroll
            for (int j = 0; j < 8; ++j) { a[mi][j] = lo[j]; a[mi][j + 8] = hi[j]; }
        }
        #pragma unroll
        for (int t = 0; t < 4; ++t) {
            int off = (wn * 64 + t * 16 + lr) * LDB + klo;
            h8 lo = *(const h8*)&ldsB[off];
            h8 hi = *(const h8*)&ldsB[off + 16];
            #pragma unroll
            for (int j = 0; j < 8; ++j) { b[t][j] = lo[j]; b[t][j + 8] = hi[j]; }
        }

        // ---- 8x v_wmma_f32_16x16x32_f16 ----
        #pragma unroll
        for (int mi = 0; mi < 2; ++mi)
            #pragma unroll
            for (int t = 0; t < 4; ++t)
                c[mi][t] = __builtin_amdgcn_wmma_f32_16x16x32_f16(
                    false, a[mi], false, b[t], (short)0, c[mi][t], false, false);

        __syncthreads();
    }

    // ---- fused epilogue: +b_ft, SCReLU, * W_out, reduce cols, atomicAdd ----
    float bias[4], wv[4];
    #pragma unroll
    for (int t = 0; t < 4; ++t) {
        int o = n0 + wn * 64 + t * 16 + lr;
        bias[t] = b_ft[o];
        wv[t]   = W_out[p * FT + o];
    }
    #pragma unroll
    for (int mi = 0; mi < 2; ++mi) {
        #pragma unroll
        for (int i = 0; i < 8; ++i) {
            float s = 0.0f;
            #pragma unroll
            for (int t = 0; t < 4; ++t) {
                float x = c[mi][t][i] + bias[t];
                x = fminf(fmaxf(x, 0.0f), 1.0f);
                s += x * x * wv[t];
            }
            // reduce the 16 lanes of this half-wave (each holds one column)
            s += __shfl_xor(s, 1, 32);
            s += __shfl_xor(s, 2, 32);
            s += __shfl_xor(s, 4, 32);
            s += __shfl_xor(s, 8, 32);
            if (lr == 0) {
                int row = m0 + wm * 32 + mi * 16 + i + lhalf * 8;
                atomicAdd(&out_acc[row], s);
            }
        }
    }
}

__global__ __launch_bounds__(256) void zero_out_kernel(float* __restrict__ out)
{
    out[blockIdx.x * 256 + threadIdx.x] = 0.0f;
}

__global__ __launch_bounds__(256) void sigmoid_kernel(float* __restrict__ out,
                                                      const float* __restrict__ b_out)
{
    int i = blockIdx.x * 256 + threadIdx.x;
    float v = out[i] + b_out[0];
    out[i] = 1.0f / (1.0f + __expf(-v));
}

extern "C" void kernel_launch(void* const* d_in, const int* in_sizes, int n_in,
                              void* d_out, int out_size, void* d_ws, size_t ws_size,
                              hipStream_t stream)
{
    (void)in_sizes; (void)n_in; (void)d_ws; (void)ws_size; (void)out_size;

    const float* board_stm  = (const float*)d_in[0];
    const float* board_nstm = (const float*)d_in[1];
    const float* W_ft       = (const float*)d_in[2];
    const float* b_ft       = (const float*)d_in[3];
    const float* W_out      = (const float*)d_in[4];
    const float* b_out      = (const float*)d_in[5];
    float* out = (float*)d_out;

    zero_out_kernel<<<BATCH / 256, 256, 0, stream>>>(out);

    dim3 grid(FT / BN, BATCH / BM, 2);
    ft_gemm_fused<<<grid, 256, 0, stream>>>(board_stm, board_nstm, W_ft, b_ft, W_out, out);

    sigmoid_kernel<<<BATCH / 256, 256, 0, stream>>>(out, b_out);
}
